// MeOViT_7911329759615
// MI455X (gfx1250) — compile-verified, matrix operand order checked by reading, not run on hardware
//
#include <hip/hip_runtime.h>
#include <hip/hip_bf16.h>
#include <stdint.h>

typedef __attribute__((ext_vector_type(8)))  _Float16 v8h;
typedef __attribute__((ext_vector_type(16))) _Float16 v16h;
typedef __attribute__((ext_vector_type(8)))  float    v8f;
typedef __attribute__((ext_vector_type(4)))  unsigned int v4u;
typedef __attribute__((ext_vector_type(8)))  int      v8i;
typedef __attribute__((ext_vector_type(4)))  int      v4i;

union F16Frag { v16h v; v8h h[2]; };

// ---- dims ----
static constexpr int Bsz = 32, Dm = 768, NH = 12, Lnum = 6, FF = 3072;
static constexpr int NP = 196, Sq = 197, SP = 224, HD = 64;
static constexpr int KEEP = 98, NE = 8, HE = 512;

__device__ __forceinline__ float gelu_f(float x) {
    return 0.5f * x * (1.f + erff(x * 0.70710678118f));
}

// ---------------------------------------------------------------------------
// TDM: DMA one 2-D f16 tile (tileRows x 32) from global [ldK-strided] to LDS.
// D# per CDNA5 ISA ch8: group0 = {flags, lds_addr, global_addr, type=2},
// group1 = {data_size=2B, tensor dims (for OOB zero-fill), tile dims, stride}.
// Issued by one wave; completion tracked with TENSORcnt.
// ---------------------------------------------------------------------------
__device__ __forceinline__ void tdm_load_tile_f16(
    const _Float16* gbase, unsigned ldsAddr, int rowsRemain, int kRemain,
    int ldK, int tileRows)
{
    unsigned long long ga = (unsigned long long)(uintptr_t)gbase;
    unsigned d0 = (unsigned)kRemain;          // tensor_dim0 (elements, K)
    unsigned d1 = (unsigned)rowsRemain;       // tensor_dim1 (rows, N)
    unsigned long long st = (unsigned long long)(unsigned)ldK; // dim0 stride

    v4u g0;
    g0[0] = 1u;                                            // count=1, user D#
    g0[1] = ldsAddr;                                       // lds_addr
    g0[2] = (unsigned)(ga & 0xffffffffu);                  // global_addr lo
    g0[3] = (unsigned)((ga >> 32) & 0x01ffffffu) | (2u << 30); // addr hi | type=2

    v8i g1;
    g1[0] = (int)(1u << 16);                               // data_size=1 (2B)
    g1[1] = (int)((d0 & 0xffffu) << 16);                   // tensor_dim0[15:0]
    g1[2] = (int)(((d0 >> 16) & 0xffffu) | ((d1 & 0xffffu) << 16));
    g1[3] = (int)(((d1 >> 16) & 0xffffu) | (32u << 16));   // tile_dim0 = 32
    g1[4] = (int)((unsigned)tileRows & 0xffffu);           // tile_dim1, tile_dim2=0
    g1[5] = (int)(unsigned)(st & 0xffffffffu);             // dim0 stride lo
    g1[6] = (int)((st >> 32) & 0xffffu);                   // dim0 stride hi
    g1[7] = 0;

    v4i z4 = {0, 0, 0, 0};
    v8i z8 = {0, 0, 0, 0, 0, 0, 0, 0};
    __builtin_amdgcn_tensor_load_to_lds(g0, g1, z4, z4, z8, 0);
}

// ============================================================================
// Batched WMMA GEMM:  C[M,N] = A[M,K] * Bw[N,K]^T (+ bias[N])
// A,Bw f16 row-major; C f32 row-major (ldc=N). Strides are element counts.
// 256 thr = 8 waves; block tile 128x128x32; wave tile 64x32 -> 8 WMMA/iter.
// A tile: per-lane async global->LDS (ASYNCcnt); B tile: TDM (TENSORcnt).
// Double-buffered LDS, one workgroup barrier per K tile.
// ============================================================================
__global__ __launch_bounds__(256) void gemm_wmma_f16(
    const _Float16* __restrict__ A, const _Float16* __restrict__ Bw,
    const float* __restrict__ bias, float* __restrict__ C,
    int M, int N, int K, long sA, long sB, long sBias, long sC)
{
    constexpr int TM = 128, TN = 128, TK = 32;
    __shared__ __align__(16) _Float16 As[2][TM][TK];
    __shared__ __align__(16) _Float16 Bs[2][TN][TK];

    const int bz = blockIdx.z;
    A  += sA * bz;
    Bw += sB * bz;
    C  += sC * bz;
    const float* bp = bias ? bias + sBias * bz : nullptr;

    const int m0 = blockIdx.y * TM, n0 = blockIdx.x * TN;
    const int tid = threadIdx.x, lane = tid & 31, wave = tid >> 5;
    const int wm = wave & 1, wn = wave >> 1;        // 2 (M) x 4 (N) wave grid
    const int mfrag = lane & 15, khalf = lane >> 4; // ISA 16-bit frag layout

    v8f acc[4][2] = {};

    // A staging: 128x32 halves = 512 x b128 chunks, 2 per thread, async->LDS
    auto stageA = [&](int buf, int kk) {
#pragma unroll
        for (int c = tid; c < (TM * TK) / 8; c += 256) {
            const int r = c >> 2, col = (c & 3) * 8;
            int gm = m0 + r; if (gm >= M) gm = M - 1;
            const _Float16* gp = A + (long)gm * K + kk + col;
            unsigned lds = (unsigned)(uintptr_t)(&As[buf][r][col]);
            asm volatile("global_load_async_to_lds_b128 %0, %1, off"
                         :: "v"(lds), "v"(gp) : "memory");
        }
    };
    // B staging: one TDM descriptor moves the whole 128x32 weight tile.
    auto stageB = [&](int buf, int kk) {
        if (wave == 0)
            tdm_load_tile_f16(Bw + (long)n0 * K + kk,
                              (unsigned)(uintptr_t)(&Bs[buf][0][0]),
                              N - n0, K - kk, K, TN);
    };
    auto waitCopies = [&]() {
        asm volatile("s_wait_asynccnt 0x0" ::: "memory");
        if (wave == 0) __builtin_amdgcn_s_wait_tensorcnt((short)0);
        __syncthreads();
    };

    stageA(0, 0);
    stageB(0, 0);
    waitCopies();

    int buf = 0;
    for (int kk = 0; kk < K; kk += TK) {
        const int nk = kk + TK;
        if (nk < K) { stageA(buf ^ 1, nk); stageB(buf ^ 1, nk); }

        F16Frag a[4], b[2];
#pragma unroll
        for (int i = 0; i < 4; ++i) {
            a[i].h[0] = *(const v8h*)&As[buf][wm * 64 + i * 16 + mfrag][khalf * 8];
            a[i].h[1] = *(const v8h*)&As[buf][wm * 64 + i * 16 + mfrag][16 + khalf * 8];
        }
#pragma unroll
        for (int j = 0; j < 2; ++j) {
            b[j].h[0] = *(const v8h*)&Bs[buf][wn * 32 + j * 16 + mfrag][khalf * 8];
            b[j].h[1] = *(const v8h*)&Bs[buf][wn * 32 + j * 16 + mfrag][16 + khalf * 8];
        }
#pragma unroll
        for (int i = 0; i < 4; ++i)
#pragma unroll
            for (int j = 0; j < 2; ++j)
                acc[i][j] = __builtin_amdgcn_wmma_f32_16x16x32_f16(
                    false, a[i].v, false, b[j].v, (short)0, acc[i][j], false, false);

        waitCopies();
        buf ^= 1;
    }

    // C/D layout: VGPR r -> m = r + (lane>>4)*8 ; n = lane&15
#pragma unroll
    for (int i = 0; i < 4; ++i) {
        const int mb = m0 + wm * 64 + i * 16 + khalf * 8;
#pragma unroll
        for (int j = 0; j < 2; ++j) {
            const int nc = n0 + wn * 32 + j * 16 + mfrag;
            if (nc < N) {
                const float bv = bp ? bp[nc] : 0.f;
#pragma unroll
                for (int r = 0; r < 8; ++r) {
                    const int gm = mb + r;
                    if (gm < M) C[(long)gm * N + nc] = acc[i][j][r] + bv;
                }
            }
        }
    }
}

// ============================================================================
// Elementwise / helper kernels
// ============================================================================
__global__ void k_f32_to_f16(const float* __restrict__ s, _Float16* __restrict__ d,
                             long n, int act)
{
    long i = (long)blockIdx.x * blockDim.x + threadIdx.x;
    if (i >= n) return;
    float v = s[i];
    if (act == 1) v = fmaxf(v, 0.f);
    else if (act == 2) v = gelu_f(v);
    d[i] = (_Float16)v;
}

// x (B,3,224,224) -> Apat [B*196][768], col = c*256 + dy*16 + dx
__global__ void k_im2col(const float* __restrict__ x, _Float16* __restrict__ Apat)
{
    long i = (long)blockIdx.x * blockDim.x + threadIdx.x;
    if (i >= (long)Bsz * NP * Dm) return;
    int col = (int)(i % Dm);
    long row = i / Dm;
    int b = (int)(row / NP), n = (int)(row % NP);
    int py = n / 14, px = n % 14;
    int c = col / 256, rem = col % 256, dy = rem / 16, dx = rem % 16;
    float v = x[(((long)b * 3 + c) * 224 + (py * 16 + dy)) * 224 + (px * 16 + dx)];
    Apat[i] = (_Float16)v;
}

// conv out (bias already added in GEMM) + cls + pos -> h f32 [B][197][768]
__global__ void k_assemble(const float* __restrict__ conv, const float* __restrict__ cls,
                           const float* __restrict__ pos, float* __restrict__ h)
{
    long i = (long)blockIdx.x * blockDim.x + threadIdx.x;
    if (i >= (long)Bsz * Sq * Dm) return;
    int d = (int)(i % Dm);
    long r = i / Dm;
    int s = (int)(r % Sq);
    float v;
    if (s == 0) v = cls[d] + pos[d];
    else        v = conv[((long)(r / Sq) * NP + (s - 1)) * Dm + d] + pos[(long)s * Dm + d];
    h[i] = v;
}

// qkv f32 [B*197][2304] -> q16,k16 [B*NH][224][64] (q scaled 1/8), v16t [B*NH][64][224]
__global__ void k_pack_qkv(const float* __restrict__ qkv, _Float16* __restrict__ q16,
                           _Float16* __restrict__ k16, _Float16* __restrict__ v16t)
{
    long i = (long)blockIdx.x * blockDim.x + threadIdx.x;
    if (i >= (long)Bsz * NH * SP * HD) return;
    int d = (int)(i % HD);
    long r = i / HD;
    int s = (int)(r % SP);
    long bh = r / SP;
    int b = (int)(bh / NH), hh = (int)(bh % NH);
    float qv = 0.f, kv = 0.f, vv = 0.f;
    if (s < Sq) {
        long base = ((long)(b * Sq + s)) * (3 * Dm) + hh * HD + d;
        qv = qkv[base] * 0.125f;       // 1/sqrt(64)
        kv = qkv[base + Dm];
        vv = qkv[base + 2 * Dm];
    }
    q16[bh * (long)SP * HD + (long)s * HD + d] = (_Float16)qv;
    k16[bh * (long)SP * HD + (long)s * HD + d] = (_Float16)kv;
    v16t[bh * (long)HD * SP + (long)d * SP + s] = (_Float16)vv;
}

// softmax over cols<197 of scores f32 [row=bh*224+m][224] -> att f16 (pads zero)
__global__ __launch_bounds__(256) void k_softmax(const float* __restrict__ scores,
                                                 _Float16* __restrict__ att)
{
    const int row = blockIdx.x;
    const int m = row % SP;
    const float* src = scores + (long)row * SP;
    _Float16* dst = att + (long)row * SP;
    const int tid = threadIdx.x;
    __shared__ float red[8];
    __shared__ float smax, ssum;

    float mx = -1e30f;
    if (m < Sq)
        for (int j = tid; j < Sq; j += 256) mx = fmaxf(mx, src[j]);
    for (int o = 16; o; o >>= 1) mx = fmaxf(mx, __shfl_down(mx, o, 32));
    if ((tid & 31) == 0) red[tid >> 5] = mx;
    __syncthreads();
    if (tid == 0) { float v = red[0]; for (int i = 1; i < 8; ++i) v = fmaxf(v, red[i]); smax = v; }
    __syncthreads();

    float sum = 0.f;
    if (m < Sq)
        for (int j = tid; j < Sq; j += 256) sum += __expf(src[j] - smax);
    for (int o = 16; o; o >>= 1) sum += __shfl_down(sum, o, 32);
    if ((tid & 31) == 0) red[tid >> 5] = sum;
    __syncthreads();
    if (tid == 0) { float v = 0.f; for (int i = 0; i < 8; ++i) v += red[i]; ssum = v; }
    __syncthreads();

    const float inv = (m < Sq && ssum > 0.f) ? 1.f / ssum : 0.f;
    for (int j = tid; j < SP; j += 256) {
        float v = (m < Sq && j < Sq) ? __expf(src[j] - smax) * inv : 0.f;
        dst[j] = (_Float16)v;
    }
}

// O f32 [B*NH][224][64] -> o16 [B*197][768]
__global__ void k_merge_o(const float* __restrict__ O, _Float16* __restrict__ o16)
{
    long i = (long)blockIdx.x * blockDim.x + threadIdx.x;
    if (i >= (long)Bsz * Sq * Dm) return;
    int col = (int)(i % Dm);
    long r = i / Dm;
    int b = (int)(r / Sq), s = (int)(r % Sq);
    int hh = col / HD, d = col % HD;
    o16[i] = (_Float16)O[(((long)(b * NH + hh)) * SP + s) * HD + d];
}

// h = LN(h + delta)*g + b  (f32), also emits f16 copy for next GEMM
__global__ __launch_bounds__(256) void k_residual_ln(float* __restrict__ h,
                                                     const float* __restrict__ delta,
                                                     const float* __restrict__ g,
                                                     const float* __restrict__ bta,
                                                     _Float16* __restrict__ h16)
{
    const int row = blockIdx.x;
    float* hp = h + (long)row * Dm;
    const float* dp = delta + (long)row * Dm;
    const int tid = threadIdx.x;
    __shared__ float red[8];
    __shared__ float mean_s, rstd_s;

    float x0 = hp[tid] + dp[tid];
    float x1 = hp[tid + 256] + dp[tid + 256];
    float x2 = hp[tid + 512] + dp[tid + 512];

    float s = x0 + x1 + x2;
    for (int o = 16; o; o >>= 1) s += __shfl_down(s, o, 32);
    if ((tid & 31) == 0) red[tid >> 5] = s;
    __syncthreads();
    if (tid == 0) { float t = 0.f; for (int i = 0; i < 8; ++i) t += red[i]; mean_s = t / (float)Dm; }
    __syncthreads();
    const float mu = mean_s;
    float d0 = x0 - mu, d1 = x1 - mu, d2 = x2 - mu;
    float sq = d0 * d0 + d1 * d1 + d2 * d2;
    for (int o = 16; o; o >>= 1) sq += __shfl_down(sq, o, 32);
    if ((tid & 31) == 0) red[tid >> 5] = sq;
    __syncthreads();
    if (tid == 0) { float t = 0.f; for (int i = 0; i < 8; ++i) t += red[i];
                    rstd_s = rsqrtf(t / (float)Dm + 1e-5f); }
    __syncthreads();
    const float rs = rstd_s;

    float y0 = d0 * rs * g[tid] + bta[tid];
    float y1 = d1 * rs * g[tid + 256] + bta[tid + 256];
    float y2 = d2 * rs * g[tid + 512] + bta[tid + 512];
    hp[tid] = y0;        hp[tid + 256] = y1;        hp[tid + 512] = y2;
    _Float16* hq = h16 + (long)row * Dm;
    hq[tid] = (_Float16)y0; hq[tid + 256] = (_Float16)y1; hq[tid + 512] = (_Float16)y2;
}

// tok16 = h16 without the cls token
__global__ void k_copy_tok16(const _Float16* __restrict__ h16, _Float16* __restrict__ tok16)
{
    long i = (long)blockIdx.x * blockDim.x + threadIdx.x;
    if (i >= (long)Bsz * NP * Dm) return;
    int d = (int)(i % Dm);
    long r = i / Dm;
    int b = (int)(r / NP), n = (int)(r % NP);
    tok16[i] = h16[((long)b * Sq + 1 + n) * Dm + d];
}

// score head: sc[row] = dot(s1[row][0:384], Ws2) + bs2   (one wave per row)
__global__ __launch_bounds__(256) void k_dot384(const _Float16* __restrict__ s1,
                                                const _Float16* __restrict__ w,
                                                const float* __restrict__ bs2,
                                                float* __restrict__ sc, int rows)
{
    int wv = blockIdx.x * 8 + (threadIdx.x >> 5);
    int lane = threadIdx.x & 31;
    if (wv >= rows) return;
    const _Float16* a = s1 + (long)wv * 384;
    float sum = 0.f;
    for (int j = lane; j < 384; j += 32) sum += (float)a[j] * (float)w[j];
    for (int o = 16; o; o >>= 1) sum += __shfl_down(sum, o, 32);
    if (lane == 0) sc[wv] = sum + bs2[0];
}

// top-98 of 196 (descending, deterministic) per batch
__global__ void k_topk98(const float* __restrict__ sc, int* __restrict__ idx)
{
    const int b = blockIdx.x;
    if (threadIdx.x != 0) return;
    float local[NP];
    for (int i = 0; i < NP; ++i) local[i] = sc[b * NP + i];
    for (int k = 0; k < KEEP; ++k) {
        int best = 0; float bv = local[0];
        for (int i = 1; i < NP; ++i) if (local[i] > bv) { bv = local[i]; best = i; }
        idx[b * KEEP + k] = best;
        local[best] = -1e30f;
    }
}

// gather selected tokens (from final f32 h) into f16
__global__ void k_gather_sel(const float* __restrict__ h, const int* __restrict__ idx,
                             _Float16* __restrict__ sel)
{
    long i = (long)blockIdx.x * blockDim.x + threadIdx.x;
    if (i >= (long)Bsz * KEEP * Dm) return;
    int d = (int)(i % Dm);
    long r = i / Dm;
    int b = (int)(r / KEEP), j = (int)(r % KEEP);
    int t = idx[b * KEEP + j];
    sel[i] = (_Float16)h[((long)b * Sq + 1 + t) * Dm + d];
}

// gate logits (8 dots of 768) + top-2 softmax -> dense gates[row][8]
__global__ __launch_bounds__(256) void k_gate(const _Float16* __restrict__ sel,
                                              const _Float16* __restrict__ Wg,
                                              const float* __restrict__ bg,
                                              float* __restrict__ gates, int rows)
{
    int wv = blockIdx.x * 8 + (threadIdx.x >> 5);
    int lane = threadIdx.x & 31;
    if (wv >= rows) return;
    float logit[NE];
    for (int e = 0; e < NE; ++e) {
        float s = 0.f;
        for (int j = lane; j < Dm; j += 32)
            s += (float)sel[(long)wv * Dm + j] * (float)Wg[(long)e * Dm + j];
        for (int o = 16; o; o >>= 1) s += __shfl_down(s, o, 32);
        s = __shfl(s, 0, 32);
        logit[e] = s + bg[e];
    }
    if (lane == 0) {
        int i0 = 0;
        for (int e = 1; e < NE; ++e) if (logit[e] > logit[i0]) i0 = e;
        int i1 = (i0 == 0) ? 1 : 0;
        for (int e = 0; e < NE; ++e) if (e != i0 && logit[e] > logit[i1]) i1 = e;
        float m = logit[i0];
        float e0 = __expf(logit[i0] - m), e1 = __expf(logit[i1] - m);
        float inv = 1.f / (e0 + e1);
        for (int e = 0; e < NE; ++e) gates[(long)wv * NE + e] = 0.f;
        gates[(long)wv * NE + i0] = e0 * inv;
        gates[(long)wv * NE + i1] = e1 * inv;
    }
}

// out[b][hc] = sum_{n,e} gates[(b*98+n)][e] * h3[e][(b*98+n)][hc]
__global__ __launch_bounds__(512) void k_moe_reduce(const float* __restrict__ h3,
                                                    const float* __restrict__ gates,
                                                    float* __restrict__ out)
{
    const int b = blockIdx.x;
    const int hc = threadIdx.x;
    float acc = 0.f;
    for (int n = 0; n < KEEP; ++n) {
        long row = (long)b * KEEP + n;
        for (int e = 0; e < NE; ++e) {
            float g = gates[row * NE + e];
            if (g != 0.f)
                acc += g * h3[((long)e * (Bsz * KEEP) + row) * HE + hc];
        }
    }
    out[(long)b * HE + hc] = acc;
}

// ============================================================================
// Host orchestration
// ============================================================================
static inline int cdiv(long a, long b) { return (int)((a + b - 1) / b); }

extern "C" void kernel_launch(void* const* d_in, const int* in_sizes, int n_in,
                              void* d_out, int out_size, void* d_ws, size_t ws_size,
                              hipStream_t stream)
{
    (void)in_sizes; (void)n_in; (void)out_size; (void)ws_size;

    const float* x    = (const float*)d_in[0];
    const float* Wp   = (const float*)d_in[1];
    const float* bp   = (const float*)d_in[2];
    const float* cls  = (const float*)d_in[3];
    const float* pos  = (const float*)d_in[4];
    const float* Wqkv = (const float*)d_in[5];
    const float* bqkv = (const float*)d_in[6];
    const float* Wo   = (const float*)d_in[7];
    const float* bo   = (const float*)d_in[8];
    const float* g1   = (const float*)d_in[9];
    const float* b1   = (const float*)d_in[10];
    const float* W1   = (const float*)d_in[11];
    const float* bf1  = (const float*)d_in[12];
    const float* W2   = (const float*)d_in[13];
    const float* bf2  = (const float*)d_in[14];
    const float* g2   = (const float*)d_in[15];
    const float* b2   = (const float*)d_in[16];
    const float* Ws1  = (const float*)d_in[17];
    const float* bs1  = (const float*)d_in[18];
    const float* Ws2  = (const float*)d_in[19];
    const float* bs2  = (const float*)d_in[20];
    const float* Wg   = (const float*)d_in[21];
    const float* bg   = (const float*)d_in[22];
    const float* We1  = (const float*)d_in[23];
    const float* be1  = (const float*)d_in[24];
    const float* We2  = (const float*)d_in[25];
    const float* be2  = (const float*)d_in[26];
    const float* We3  = (const float*)d_in[27];
    const float* be3  = (const float*)d_in[28];
    float* out = (float*)d_out;

    size_t off = 0;
    auto a16 = [&](long elems) -> _Float16* {
        _Float16* p = (_Float16*)((char*)d_ws + off);
        off += ((size_t)elems * 2 + 255) & ~(size_t)255; return p;
    };
    auto a32 = [&](long elems) -> float* {
        float* p = (float*)((char*)d_ws + off);
        off += ((size_t)elems * 4 + 255) & ~(size_t)255; return p;
    };
    auto aI = [&](long elems) -> int* {
        int* p = (int*)((char*)d_ws + off);
        off += ((size_t)elems * 4 + 255) & ~(size_t)255; return p;
    };

    const long BS = (long)Bsz * Sq;          // 6304
    const long BNP = (long)Bsz * NP;         // 6272
    const long BH = (long)Bsz * NH;          // 384
    const long NSEL = (long)Bsz * KEEP;      // 3136

    // ---- f16 weight mirrors ----
    _Float16* Wp16   = a16((long)Dm * Dm);
    _Float16* Wqkv16 = a16((long)Lnum * 3 * Dm * Dm);
    _Float16* Wo16   = a16((long)Lnum * Dm * Dm);
    _Float16* W116   = a16((long)Lnum * FF * Dm);
    _Float16* W216   = a16((long)Lnum * Dm * FF);
    _Float16* Ws116  = a16((long)(Dm / 2) * Dm);
    _Float16* Ws216  = a16(Dm / 2);
    _Float16* Wg16   = a16((long)NE * Dm);
    _Float16* We116  = a16((long)NE * HE * Dm);
    _Float16* We216  = a16((long)NE * HE * HE);
    _Float16* We316  = a16((long)NE * HE * HE);

    // ---- activations ----
    _Float16* Apat16 = a16(BNP * Dm);
    float*    conv32 = a32(BNP * Dm);
    float*    h32    = a32(BS * Dm);
    _Float16* h16    = a16(BS * Dm);
    float*    qkv32  = a32(BS * 3 * Dm);
    _Float16* q16    = a16(BH * SP * HD);
    _Float16* k16    = a16(BH * SP * HD);
    _Float16* v16t   = a16(BH * HD * SP);
    float*    sc32b  = a32(BH * SP * SP);    // attention scores
    _Float16* att16  = a16(BH * SP * SP);
    float*    O32    = a32(BH * SP * HD);
    _Float16* o16    = a16(BS * Dm);
    float*    delta  = a32(BS * Dm);
    float*    ff1_32 = a32(BS * FF);
    _Float16* ff1_16 = a16(BS * FF);
    _Float16* tok16  = a16(BNP * Dm);
    float*    s1_32  = a32(BNP * (Dm / 2));
    _Float16* s1_16  = a16(BNP * (Dm / 2));
    float*    scT    = a32(BNP);
    int*      idxb   = aI((long)Bsz * KEEP);
    _Float16* sel16  = a16(NSEL * Dm);
    float*    gates  = a32(NSEL * NE);
    float*    h1_32  = a32((long)NE * NSEL * HE);
    _Float16* h1_16  = a16((long)NE * NSEL * HE);
    float*    h2_32  = a32((long)NE * NSEL * HE);
    _Float16* h2_16  = a16((long)NE * NSEL * HE);
    float*    h3_32  = a32((long)NE * NSEL * HE);

    auto cvt = [&](const float* s, _Float16* d, long n, int act) {
        k_f32_to_f16<<<cdiv(n, 256), 256, 0, stream>>>(s, d, n, act);
    };

    // ---- convert weights to f16 ----
    cvt(Wp,   Wp16,   (long)Dm * Dm, 0);
    cvt(Wqkv, Wqkv16, (long)Lnum * 3 * Dm * Dm, 0);
    cvt(Wo,   Wo16,   (long)Lnum * Dm * Dm, 0);
    cvt(W1,   W116,   (long)Lnum * FF * Dm, 0);
    cvt(W2,   W216,   (long)Lnum * Dm * FF, 0);
    cvt(Ws1,  Ws116,  (long)(Dm / 2) * Dm, 0);
    cvt(Ws2,  Ws216,  Dm / 2, 0);
    cvt(Wg,   Wg16,   (long)NE * Dm, 0);
    cvt(We1,  We116,  (long)NE * HE * Dm, 0);
    cvt(We2,  We216,  (long)NE * HE * HE, 0);
    cvt(We3,  We316,  (long)NE * HE * HE, 0);

    // ---- patch embed ----
    k_im2col<<<cdiv(BNP * Dm, 256), 256, 0, stream>>>(x, Apat16);
    {
        dim3 g(cdiv(Dm, 128), cdiv(BNP, 128), 1);
        gemm_wmma_f16<<<g, 256, 0, stream>>>(Apat16, Wp16, bp, conv32,
                                             (int)BNP, Dm, Dm, 0, 0, 0, 0);
    }
    k_assemble<<<cdiv(BS * Dm, 256), 256, 0, stream>>>(conv32, cls, pos, h32);
    cvt(h32, h16, BS * Dm, 0);

    // ---- transformer layers ----
    for (int i = 0; i < Lnum; ++i) {
        {
            dim3 g(cdiv(3 * Dm, 128), cdiv(BS, 128), 1);
            gemm_wmma_f16<<<g, 256, 0, stream>>>(h16, Wqkv16 + (long)i * 3 * Dm * Dm,
                                                 bqkv + (long)i * 3 * Dm, qkv32,
                                                 (int)BS, 3 * Dm, Dm, 0, 0, 0, 0);
        }
        k_pack_qkv<<<cdiv(BH * SP * HD, 256), 256, 0, stream>>>(qkv32, q16, k16, v16t);
        {
            dim3 g(cdiv(SP, 128), cdiv(SP, 128), (unsigned)BH);
            gemm_wmma_f16<<<g, 256, 0, stream>>>(q16, k16, nullptr, sc32b,
                                                 SP, SP, HD,
                                                 (long)SP * HD, (long)SP * HD, 0,
                                                 (long)SP * SP);
        }
        k_softmax<<<(unsigned)(BH * SP), 256, 0, stream>>>(sc32b, att16);
        {
            dim3 g(cdiv(HD, 128), cdiv(SP, 128), (unsigned)BH);
            gemm_wmma_f16<<<g, 256, 0, stream>>>(att16, v16t, nullptr, O32,
                                                 SP, HD, SP,
                                                 (long)SP * SP, (long)HD * SP, 0,
                                                 (long)SP * HD);
        }
        k_merge_o<<<cdiv(BS * Dm, 256), 256, 0, stream>>>(O32, o16);
        {
            dim3 g(cdiv(Dm, 128), cdiv(BS, 128), 1);
            gemm_wmma_f16<<<g, 256, 0, stream>>>(o16, Wo16 + (long)i * Dm * Dm,
                                                 bo + (long)i * Dm, delta,
                                                 (int)BS, Dm, Dm, 0, 0, 0, 0);
        }
        k_residual_ln<<<(unsigned)BS, 256, 0, stream>>>(h32, delta,
                                                        g1 + (long)i * Dm, b1 + (long)i * Dm, h16);
        {
            dim3 g(cdiv(FF, 128), cdiv(BS, 128), 1);
            gemm_wmma_f16<<<g, 256, 0, stream>>>(h16, W116 + (long)i * FF * Dm,
                                                 bf1 + (long)i * FF, ff1_32,
                                                 (int)BS, FF, Dm, 0, 0, 0, 0);
        }
        cvt(ff1_32, ff1_16, BS * FF, 1);  // relu
        {
            dim3 g(cdiv(Dm, 128), cdiv(BS, 128), 1);
            gemm_wmma_f16<<<g, 256, 0, stream>>>(ff1_16, W216 + (long)i * Dm * FF,
                                                 bf2 + (long)i * Dm, delta,
                                                 (int)BS, Dm, FF, 0, 0, 0, 0);
        }
        k_residual_ln<<<(unsigned)BS, 256, 0, stream>>>(h32, delta,
                                                        g2 + (long)i * Dm, b2 + (long)i * Dm, h16);
    }

    // ---- token scoring + top-k selection ----
    k_copy_tok16<<<cdiv(BNP * Dm, 256), 256, 0, stream>>>(h16, tok16);
    {
        dim3 g(cdiv(Dm / 2, 128), cdiv(BNP, 128), 1);
        gemm_wmma_f16<<<g, 256, 0, stream>>>(tok16, Ws116, bs1, s1_32,
                                             (int)BNP, Dm / 2, Dm, 0, 0, 0, 0);
    }
    cvt(s1_32, s1_16, BNP * (Dm / 2), 1);  // relu
    k_dot384<<<cdiv(BNP, 8), 256, 0, stream>>>(s1_16, Ws216, bs2, scT, (int)BNP);
    k_topk98<<<Bsz, 32, 0, stream>>>(scT, idxb);
    k_gather_sel<<<cdiv(NSEL * Dm, 256), 256, 0, stream>>>(h32, idxb, sel16);
    k_gate<<<cdiv(NSEL, 8), 256, 0, stream>>>(sel16, Wg16, bg, gates, (int)NSEL);

    // ---- MoE experts (batched over 8 experts) ----
    {
        dim3 g(cdiv(HE, 128), cdiv(NSEL, 128), NE);
        gemm_wmma_f16<<<g, 256, 0, stream>>>(sel16, We116, be1, h1_32,
                                             (int)NSEL, HE, Dm,
                                             0, (long)HE * Dm, HE, (long)NSEL * HE);
    }
    cvt(h1_32, h1_16, (long)NE * NSEL * HE, 2);  // gelu
    {
        dim3 g(cdiv(HE, 128), cdiv(NSEL, 128), NE);
        gemm_wmma_f16<<<g, 256, 0, stream>>>(h1_16, We216, be2, h2_32,
                                             (int)NSEL, HE, HE,
                                             (long)NSEL * HE, (long)HE * HE, HE,
                                             (long)NSEL * HE);
    }
    cvt(h2_32, h2_16, (long)NE * NSEL * HE, 2);  // gelu
    {
        dim3 g(cdiv(HE, 128), cdiv(NSEL, 128), NE);
        gemm_wmma_f16<<<g, 256, 0, stream>>>(h2_16, We316, be3, h3_32,
                                             (int)NSEL, HE, HE,
                                             (long)NSEL * HE, (long)HE * HE, HE,
                                             (long)NSEL * HE);
    }
    k_moe_reduce<<<Bsz, HE, 0, stream>>>(h3_32, gates, out);
}
